// CausalSelfAttention_2963527434746
// MI455X (gfx1250) — matrix operand order repaired and checked by reading, not compiled, not run on hardware
//
#include <hip/hip_runtime.h>

// Causal self-attention forward for MI455X (gfx1250, wave32, WMMA).
// B=2, T=2048, C=1024, H=16, HD=64. ~70 GFLOP -> compute bound -> f16 WMMA.
// GEMMs use 64x64 register tiles/wave (16 WMMA per 8KB loaded, ~33 FLOP/B from L2);
// attention uses 32-query waves (16 WMMA per 8KB of K+V).
#define B_  2
#define T_  2048
#define C_  1024
#define H_  16
#define HD_ 64
#define M_TOK (B_*T_)          // 4096 tokens

typedef __attribute__((ext_vector_type(16))) _Float16 v16h;
typedef __attribute__((ext_vector_type(8)))  _Float16 v8h;
typedef __attribute__((ext_vector_type(8)))  float    v8f;

union HF16 { v16h v; v8h h[2]; };

__device__ __forceinline__ v8f wmma_f16f32(v16h a, v16h b, v8f c) {
  // (neg_a, A, neg_b, B, c_mod, C, reuse_a, reuse_b)
  return __builtin_amdgcn_wmma_f32_16x16x32_f16(false, a, false, b, (short)0, c,
                                                false, false);
}

// WGP-scope prefetch (scope 0 = default = pull into all cache levels on miss).
// __builtin_prefetch can only reach SE/DEV/SYS scopes, which skip the WGP$.
__device__ __forceinline__ void prefetch_wgp(const void* p) {
  asm volatile("global_prefetch_b8 %0, off" :: "v"(p));
}

// A fragment: 16(M) x 32(K) from row-major [.., ld] f16 source.
// lane row = lane&15; elems 0..7 -> K=(lane>>4)*8+e ; elems 8..15 -> K=16+(lane>>4)*8+(e-8)
__device__ __forceinline__ v16h load_afrag(const _Float16* base, int ld, int lane) {
  int row = lane & 15, g = lane >> 4;
  const _Float16* p = base + (size_t)row * ld + g * 8;
  HF16 u;
  u.h[0] = *(const v8h*)(p);
  u.h[1] = *(const v8h*)(p + 16);
  return u.v;
}

// B fragment: 32(K) x 16(N) taken from an N-major source ("Bt": rows=N, cols=K, ld).
// lane col = lane&15; elems 0..7 -> K=(lane>>4)*8+e ; elems 8..15 -> K=16+(lane>>4)*8+(e-8)
__device__ __forceinline__ v16h load_bfrag(const _Float16* base, int ld, int lane) {
  int n = lane & 15, g = lane >> 4;
  const _Float16* p = base + (size_t)n * ld + g * 8;
  HF16 u;
  u.h[0] = *(const v8h*)(p);
  u.h[1] = *(const v8h*)(p + 16);
  return u.v;
}

// ---------------- conversion kernels ----------------
__global__ void cvt_f32_to_f16(const float* __restrict__ in, _Float16* __restrict__ out, int n) {
  int i = blockIdx.x * blockDim.x + threadIdx.x;
  if (i < n) out[i] = (_Float16)in[i];
}

// LDS-tiled transpose+convert: out[n*rows + k] = (f16) in[k*cols + n].
// Coalesced f32 reads and coalesced f16 writes; 33-wide rows avoid bank conflicts.
// Requires rows, cols multiples of 32 (1024/3072 here). Block (32,8), grid (cols/32, rows/32).
__global__ void transpose_cvt_tiled(const float* __restrict__ in, _Float16* __restrict__ out,
                                    int rows, int cols) {
  __shared__ float tile[32][33];
  int n0 = blockIdx.x * 32;
  int k0 = blockIdx.y * 32;
  int tx = threadIdx.x, ty = threadIdx.y;
#pragma unroll
  for (int i = 0; i < 4; ++i) {
    int k = k0 + ty + i * 8;
    tile[ty + i * 8][tx] = in[(size_t)k * cols + (n0 + tx)];
  }
  __syncthreads();
#pragma unroll
  for (int i = 0; i < 4; ++i) {
    int n = n0 + ty + i * 8;
    out[(size_t)n * rows + (k0 + tx)] = (_Float16)tile[tx][ty + i * 8];
  }
}

// ---------------- QKV projection GEMM (64x64 tile per wave) ----------------
// xh [M_TOK, C] f16  @  waT [3C, C] (N-major) -> scatter Q/K head-major, V transposed.
__global__ __launch_bounds__(128) void qkv_gemm_kernel(
    const _Float16* __restrict__ xh, const _Float16* __restrict__ wT,
    const float* __restrict__ bias,
    _Float16* __restrict__ Q, _Float16* __restrict__ K, _Float16* __restrict__ Vt) {
  int lane = threadIdx.x & 31;
  int wave = blockIdx.x * 4 + (threadIdx.x >> 5);
  const int ntiles = (3 * C_) / 64;                 // 48
  int tm = wave / ntiles;                           // 0..63  (64-row tiles)
  int tn = wave - tm * ntiles;
  if (tm >= M_TOK / 64) return;

  v8f zero = {};
  v8f acc[4][4];
#pragma unroll
  for (int mi = 0; mi < 4; ++mi)
#pragma unroll
    for (int j = 0; j < 4; ++j) acc[mi][j] = zero;

  const _Float16* abase = xh + (size_t)tm * 64 * C_;
  const _Float16* bbase = wT + (size_t)tn * 64 * C_;
  for (int k0 = 0; k0 < C_; k0 += 32) {
    if (k0 + 32 < C_) {   // WGP-scope prefetch of next k-step
      prefetch_wgp(abase + (size_t)(lane * 2) * C_ + k0 + 32);
      prefetch_wgp(bbase + (size_t)(lane * 2) * C_ + k0 + 32);
    }
    v16h a[4];
#pragma unroll
    for (int mi = 0; mi < 4; ++mi)
      a[mi] = load_afrag(abase + (size_t)mi * 16 * C_ + k0, C_, lane);
#pragma unroll
    for (int j = 0; j < 4; ++j) {
      v16h b = load_bfrag(bbase + (size_t)j * 16 * C_ + k0, C_, lane);
#pragma unroll
      for (int mi = 0; mi < 4; ++mi)
        acc[mi][j] = wmma_f16f32(a[mi], b, acc[mi][j]);
    }
  }

  int g = lane >> 4, ln = lane & 15;
#pragma unroll
  for (int j = 0; j < 4; ++j) {
    int col = tn * 64 + j * 16 + ln;                // feature in [0, 3C)
    float bv = bias[col];
    int which = col / C_;                           // 0=Q 1=K 2=V (uniform per j)
    int f = col - which * C_;
    int hh = f / HD_;
    int d = f - hh * HD_;
#pragma unroll
    for (int mi = 0; mi < 4; ++mi) {
#pragma unroll
      for (int v = 0; v < 8; ++v) {
        int row = tm * 64 + mi * 16 + v + 8 * g;    // token index
        _Float16 hv = (_Float16)(acc[mi][j][v] + bv);
        int bb = row / T_;
        int t = row - bb * T_;
        size_t bh = (size_t)(bb * H_ + hh);
        if (which == 0)      Q[(bh * T_ + t) * HD_ + d] = hv;
        else if (which == 1) K[(bh * T_ + t) * HD_ + d] = hv;
        else                 Vt[(bh * HD_ + d) * T_ + t] = hv;   // transposed
      }
    }
  }
}

// ---------------- flash attention (32 queries per wave) ----------------
__global__ __launch_bounds__(128) void attn_kernel(
    const _Float16* __restrict__ Q, const _Float16* __restrict__ K,
    const _Float16* __restrict__ Vt, _Float16* __restrict__ Yh) {
  __shared__ __align__(16) _Float16 pbuf[4][2][16][32];
  int w = threadIdx.x >> 5;
  int lane = threadIdx.x & 31;
  int wave = blockIdx.x * 4 + w;
  const int qtiles = T_ / 32;                       // 64
  int bh = wave / qtiles;
  int qt = wave - bh * qtiles;
  if (bh >= B_ * H_) return;
  int g = lane >> 4, ln = lane & 15;

  const _Float16* Qbh = Q + (size_t)bh * T_ * HD_;
  const _Float16* Kbh = K + (size_t)bh * T_ * HD_;
  const _Float16* Vbh = Vt + (size_t)bh * HD_ * T_;
  int q0 = qt * 32;

  // Q fragments: 2 query sub-tiles x 2 k-steps over head dim
  v16h qa[2][2];
#pragma unroll
  for (int mi = 0; mi < 2; ++mi)
#pragma unroll
    for (int ks = 0; ks < 2; ++ks)
      qa[mi][ks] = load_afrag(Qbh + (size_t)(q0 + mi * 16) * HD_ + ks * 32, HD_, lane);

  v8f zero = {};
  v8f yacc[2][4];
#pragma unroll
  for (int mi = 0; mi < 2; ++mi)
#pragma unroll
    for (int j = 0; j < 4; ++j) yacc[mi][j] = zero;
  float mrun[2][8], lrun[2][8];
#pragma unroll
  for (int mi = 0; mi < 2; ++mi)
#pragma unroll
    for (int v = 0; v < 8; ++v) { mrun[mi][v] = -3.0e38f; lrun[mi][v] = 0.0f; }

  const float scale = 0.125f;                       // 1/sqrt(64)
  int kend = q0 + 31;                               // last valid key (causal)
  for (int kb = 0; kb <= kend; kb += 32) {
    // S = Q @ K^T for keys [kb, kb+32): shared K fragments across both q sub-tiles
    v8f s[2][2];
#pragma unroll
    for (int mi = 0; mi < 2; ++mi) { s[mi][0] = zero; s[mi][1] = zero; }
#pragma unroll
    for (int ks = 0; ks < 2; ++ks) {
      v16h kf0 = load_bfrag(Kbh + (size_t)kb * HD_ + ks * 32, HD_, lane);
      v16h kf1 = load_bfrag(Kbh + (size_t)(kb + 16) * HD_ + ks * 32, HD_, lane);
#pragma unroll
      for (int mi = 0; mi < 2; ++mi) {
        s[mi][0] = wmma_f16f32(qa[mi][ks], kf0, s[mi][0]);
        s[mi][1] = wmma_f16f32(qa[mi][ks], kf1, s[mi][1]);
      }
    }

    // online softmax per query sub-tile; write P into LDS in A layout
#pragma unroll
    for (int mi = 0; mi < 2; ++mi) {
      float p0[8], p1[8];
#pragma unroll
      for (int v = 0; v < 8; ++v) {
        int qrow = q0 + mi * 16 + v + 8 * g;
        float sa = (kb + ln      <= qrow) ? s[mi][0][v] * scale : -3.0e38f;
        float sb = (kb + 16 + ln <= qrow) ? s[mi][1][v] * scale : -3.0e38f;
        float m = fmaxf(sa, sb);
        m = fmaxf(m, __shfl_xor(m, 1, 32));
        m = fmaxf(m, __shfl_xor(m, 2, 32));
        m = fmaxf(m, __shfl_xor(m, 4, 32));
        m = fmaxf(m, __shfl_xor(m, 8, 32));
        float mnew = fmaxf(mrun[mi][v], m);
        float alpha = __expf(mrun[mi][v] - mnew);
        mrun[mi][v] = mnew;
        p0[v] = __expf(sa - mnew);
        p1[v] = __expf(sb - mnew);
        float su = p0[v] + p1[v];
        su += __shfl_xor(su, 1, 32);
        su += __shfl_xor(su, 2, 32);
        su += __shfl_xor(su, 4, 32);
        su += __shfl_xor(su, 8, 32);
        lrun[mi][v] = lrun[mi][v] * alpha + su;
#pragma unroll
        for (int j = 0; j < 4; ++j) yacc[mi][j][v] *= alpha;
      }
#pragma unroll
      for (int v = 0; v < 8; ++v) {
        int r = v + 8 * g;
        pbuf[w][mi][r][ln]      = (_Float16)p0[v];
        pbuf[w][mi][r][16 + ln] = (_Float16)p1[v];
      }
    }
    asm volatile("s_wait_dscnt 0" ::: "memory");   // wave-private DS ops are in-order

    v16h pf[2];
#pragma unroll
    for (int mi = 0; mi < 2; ++mi) {
      const _Float16* pr = &pbuf[w][mi][ln][g * 8];
      HF16 u;
      u.h[0] = *(const v8h*)(pr);
      u.h[1] = *(const v8h*)(pr + 16);
      pf[mi] = u.v;
    }

    // Y += P @ V : shared V fragments across both q sub-tiles
#pragma unroll
    for (int j = 0; j < 4; ++j) {
      v16h vf = load_bfrag(Vbh + (size_t)(j * 16) * T_ + kb, T_, lane);
#pragma unroll
      for (int mi = 0; mi < 2; ++mi)
        yacc[mi][j] = wmma_f16f32(pf[mi], vf, yacc[mi][j]);
    }
  }

  int bb = bh / H_, hh = bh - bb * H_;
#pragma unroll
  for (int mi = 0; mi < 2; ++mi) {
#pragma unroll
    for (int j = 0; j < 4; ++j) {
#pragma unroll
      for (int v = 0; v < 8; ++v) {
        int row = q0 + mi * 16 + v + 8 * g;
        int d = j * 16 + ln;
        float y = yacc[mi][j][v] / lrun[mi][v];
        Yh[((size_t)(bb * T_ + row)) * C_ + hh * HD_ + d] = (_Float16)y;
      }
    }
  }
}

// ---------------- output projection GEMM (64x64 tile per wave) ----------------
__global__ __launch_bounds__(128) void proj_gemm_kernel(
    const _Float16* __restrict__ Yh, const _Float16* __restrict__ wT,
    const float* __restrict__ bias, float* __restrict__ out) {
  int lane = threadIdx.x & 31;
  int wave = blockIdx.x * 4 + (threadIdx.x >> 5);
  const int ntiles = C_ / 64;                       // 16
  int tm = wave / ntiles;                           // 0..63
  int tn = wave - tm * ntiles;
  if (tm >= M_TOK / 64) return;

  v8f zero = {};
  v8f acc[4][4];
#pragma unroll
  for (int mi = 0; mi < 4; ++mi)
#pragma unroll
    for (int j = 0; j < 4; ++j) acc[mi][j] = zero;

  const _Float16* abase = Yh + (size_t)tm * 64 * C_;
  const _Float16* bbase = wT + (size_t)tn * 64 * C_;
  for (int k0 = 0; k0 < C_; k0 += 32) {
    if (k0 + 32 < C_) {   // WGP-scope prefetch of next k-step
      prefetch_wgp(abase + (size_t)(lane * 2) * C_ + k0 + 32);
      prefetch_wgp(bbase + (size_t)(lane * 2) * C_ + k0 + 32);
    }
    v16h a[4];
#pragma unroll
    for (int mi = 0; mi < 4; ++mi)
      a[mi] = load_afrag(abase + (size_t)mi * 16 * C_ + k0, C_, lane);
#pragma unroll
    for (int j = 0; j < 4; ++j) {
      v16h b = load_bfrag(bbase + (size_t)j * 16 * C_ + k0, C_, lane);
#pragma unroll
      for (int mi = 0; mi < 4; ++mi)
        acc[mi][j] = wmma_f16f32(a[mi], b, acc[mi][j]);
    }
  }

  int g = lane >> 4, ln = lane & 15;
#pragma unroll
  for (int j = 0; j < 4; ++j) {
    int col = tn * 64 + j * 16 + ln;
    float bv = bias[col];
#pragma unroll
    for (int mi = 0; mi < 4; ++mi) {
#pragma unroll
      for (int v = 0; v < 8; ++v) {
        int row = tm * 64 + mi * 16 + v + 8 * g;
        out[(size_t)row * C_ + col] = acc[mi][j][v] + bv;
      }
    }
  }
}

// ---------------- host launcher ----------------
extern "C" void kernel_launch(void* const* d_in, const int* in_sizes, int n_in,
                              void* d_out, int out_size, void* d_ws, size_t ws_size,
                              hipStream_t stream) {
  const float* x      = (const float*)d_in[0];
  const float* w_attn = (const float*)d_in[1];
  const float* b_attn = (const float*)d_in[2];
  const float* w_proj = (const float*)d_in[3];
  const float* b_proj = (const float*)d_in[4];
  float* out = (float*)d_out;

  // Workspace layout (f16 halves). Total 40 MB; Yh aliases xh (x dead after QKV GEMM).
  _Float16* xh  = (_Float16*)d_ws;                         // [M_TOK, C]
  _Float16* waT = xh  + (size_t)M_TOK * C_;                // [3C, C]  (N-major)
  _Float16* wpT = waT + (size_t)3 * C_ * C_;               // [C, C]   (N-major)
  _Float16* Qb  = wpT + (size_t)C_ * C_;                   // [B,H,T,HD]
  _Float16* Kb  = Qb  + (size_t)M_TOK * C_;                // [B,H,T,HD]
  _Float16* Vt  = Kb  + (size_t)M_TOK * C_;                // [B,H,HD,T]
  _Float16* Yh  = xh;                                      // alias (reuse)

  {
    int n = M_TOK * C_;
    cvt_f32_to_f16<<<(n + 255) / 256, 256, 0, stream>>>(x, xh, n);
  }
  transpose_cvt_tiled<<<dim3((3 * C_) / 32, C_ / 32), dim3(32, 8), 0, stream>>>(
      w_attn, waT, C_, 3 * C_);
  transpose_cvt_tiled<<<dim3(C_ / 32, C_ / 32), dim3(32, 8), 0, stream>>>(
      w_proj, wpT, C_, C_);

  // QKV GEMM: (M_TOK/64)*(3C/64) = 64*48 = 3072 waves -> 768 blocks of 4 waves
  qkv_gemm_kernel<<<768, 128, 0, stream>>>(xh, waT, b_attn, Qb, Kb, Vt);

  // Attention: B*H * (T/32) = 32*64 = 2048 waves -> 512 blocks
  attn_kernel<<<512, 128, 0, stream>>>(Qb, Kb, Vt, Yh);

  // Output projection: 64*16 = 1024 waves -> 256 blocks
  proj_gemm_kernel<<<256, 128, 0, stream>>>(Yh, wpT, b_proj, out);
}
